// LinearAverageContinued_8873402433953
// MI455X (gfx1250) — matrix operand order repaired and checked by reading, not compiled, not run on hardware
//
#include <hip/hip_runtime.h>

// ---------------------------------------------------------------------------
// Problem constants (from the reference)
// ---------------------------------------------------------------------------
#define GROUPS 10000
#define DIM    128
#define BATCH  4096
#define NROWS  500000
#define INV_T  (1.0f / 0.07f)
#define EPS_N  1e-12f

typedef __attribute__((ext_vector_type(16))) __bf16 bf16x16;
typedef __attribute__((ext_vector_type(8)))  __bf16 bf16x8;
typedef __attribute__((ext_vector_type(8)))  float  f32x8;
typedef __attribute__((ext_vector_type(4)))  float  f32x4;
typedef __attribute__((ext_vector_type(4)))  unsigned short u16x4;

// round-to-nearest-even float -> bf16 (stored as raw u16)
__device__ __forceinline__ unsigned short f2bf(float f) {
    unsigned u = __float_as_uint(f);
    u += 0x7FFFu + ((u >> 16) & 1u);
    return (unsigned short)(u >> 16);
}

// ---------------------------------------------------------------------------
// Kernel 1: zero the segment-sum accumulators (G*D sums + G counts)
// ---------------------------------------------------------------------------
__global__ void zero_accum_kernel(float* __restrict__ buf, int n) {
    int i = blockIdx.x * blockDim.x + threadIdx.x;
    if (i < n) buf[i] = 0.0f;
}

// ---------------------------------------------------------------------------
// Kernel 2: segmented sum.  One wave32 per memory row: 32 lanes x float4
// = 128 floats.  memory is streamed once (256 MB) -> non-temporal loads;
// accumulator is 5 MB and stays hot in L2 for the f32 atomics.
// ---------------------------------------------------------------------------
__global__ __launch_bounds__(256) void segsum_kernel(
    const float* __restrict__ mem, const int* __restrict__ gid,
    float* __restrict__ sums, float* __restrict__ counts) {
    const int lane = threadIdx.x & 31;
    const int row  = (blockIdx.x * blockDim.x + threadIdx.x) >> 5;
    if (row >= NROWS) return;
    const int g = gid[row];
    f32x4 v = __builtin_nontemporal_load(
        (const f32x4*)(mem + (size_t)row * DIM) + lane);
    float* s = sums + (size_t)g * DIM + lane * 4;
    atomicAdd(s + 0, v.x);
    atomicAdd(s + 1, v.y);
    atomicAdd(s + 2, v.z);
    atomicAdd(s + 3, v.w);
    if (lane == 0) atomicAdd(counts + g, 1.0f);
}

// ---------------------------------------------------------------------------
// Kernel 3: mean + L2-normalize + bf16 pack.  One wave32 per group.
// ---------------------------------------------------------------------------
__global__ __launch_bounds__(256) void finalize_vi_kernel(
    const float* __restrict__ sums, const float* __restrict__ counts,
    unsigned short* __restrict__ vib) {
    const int lane = threadIdx.x & 31;
    const int g    = (blockIdx.x * blockDim.x + threadIdx.x) >> 5;
    if (g >= GROUPS) return;
    const float invc = 1.0f / fmaxf(counts[g], 1.0f);
    f32x4 v = *((const f32x4*)(sums + (size_t)g * DIM) + lane);
    v.x *= invc; v.y *= invc; v.z *= invc; v.w *= invc;
    float sq = v.x * v.x + v.y * v.y + v.z * v.z + v.w * v.w;
    #pragma unroll
    for (int off = 16; off > 0; off >>= 1)
        sq += __shfl_xor(sq, off, 32);
    const float scale = 1.0f / fmaxf(sqrtf(sq), EPS_N);
    u16x4 p;
    p.x = f2bf(v.x * scale);
    p.y = f2bf(v.y * scale);
    p.z = f2bf(v.z * scale);
    p.w = f2bf(v.w * scale);
    *((u16x4*)(vib + (size_t)g * DIM) + lane) = p;
}

// ---------------------------------------------------------------------------
// Kernel 4: pack x to bf16 (row-major, 4 elems/thread)
// ---------------------------------------------------------------------------
__global__ void cvt_x_kernel(const float* __restrict__ x,
                             unsigned short* __restrict__ xb, int n4) {
    int i = blockIdx.x * blockDim.x + threadIdx.x;
    if (i >= n4) return;
    f32x4 v = *((const f32x4*)x + i);
    u16x4 p;
    p.x = f2bf(v.x); p.y = f2bf(v.y); p.z = f2bf(v.z); p.w = f2bf(v.w);
    *((u16x4*)xb + i) = p;
}

// ---------------------------------------------------------------------------
// Kernel 5: out[B,G] = (x_bf16 @ vi_bf16^T) * (1/T)  via v_wmma_f32_16x16x32_bf16
//
// 8 waves / block.  Wave computes one 16-row M tile x five 16-col N tiles
// (625 N tiles = 125 strips of 5).  A frag reused across the 5 N tiles per
// K-step; operands loaded directly from global (A+B = 3.5 MB total -> L2/WGP$
// resident).  Output (164 MB) streamed with non-temporal stores.
//
// All tile addresses are expressed as ONE per-lane base pointer per stream
// plus compile-time byte offsets (B tile (nt,kb) at nt*4096 + kb*64, A chunk
// at kb*64 + {0,16}, output at r*40000 + nt*64) so every access lowers to
// base + 24-bit immediate: minimal VALU/SALU address math and maximal
// freedom for the scheduler to clause/pipeline the loads.
//
// Per-lane operand layouts used here (contiguous K, same order for A and B):
//   A (16x32 bf16): lane l -> row (l&15),  K = kb*32 + (l>>4)*16 + {0..15}
//   B (32x16 bf16): lane l -> col (l&15),  K = kb*32 + (l>>4)*16 + {0..15}
//   C/D (16x16 f32): vgpr r, lane l -> row r + (l>>4)*8, col (l&15)
// ---------------------------------------------------------------------------
__global__ __launch_bounds__(256) void gemm_wmma_kernel(
    const __bf16* __restrict__ xb, const __bf16* __restrict__ vib,
    float* __restrict__ out) {
    const int lane  = threadIdx.x & 31;
    const int wave  = threadIdx.x >> 5;
    const int mBase = (blockIdx.y * 8 + wave) * 16;   // 32 blocks.y * 8 waves = 256 M tiles
    const int nBase = blockIdx.x * 80;                // 125 blocks.x * 5 N tiles = 625
    const int half  = lane >> 4;                      // 0: lanes 0-15, 1: lanes 16-31
    const int l16   = lane & 15;

    // single per-lane base pointers; every tile access is base + immediate
    const __bf16* aBase = xb  + (size_t)(mBase + l16) * DIM + half * 16;
    const __bf16* bBase = vib + (size_t)(nBase + l16) * DIM + half * 16;

    f32x8 acc[5] = {};

    #pragma unroll
    for (int kb = 0; kb < 4; ++kb) {                  // K = 4 * 32 = 128
        bf16x8 a0 = *(const bf16x8*)(aBase + kb * 32);       // +kb*64 B
        bf16x8 a1 = *(const bf16x8*)(aBase + kb * 32 + 8);   // +kb*64+16 B
        bf16x16 a = __builtin_shufflevector(a0, a1,
            0, 1, 2, 3, 4, 5, 6, 7, 8, 9, 10, 11, 12, 13, 14, 15);

        // load all 5 B fragments for this K-step, then burst 5 WMMAs
        bf16x16 b[5];
        #pragma unroll
        for (int nt = 0; nt < 5; ++nt)
            b[nt] = *(const bf16x16*)(bBase + nt * (16 * DIM) + kb * 32);

        #pragma unroll
        for (int nt = 0; nt < 5; ++nt)
            acc[nt] = __builtin_amdgcn_wmma_f32_16x16x32_bf16(
                /*neg_a=*/false, a, /*neg_b=*/false, b[nt],
                /*c_mod=*/(short)0, acc[nt],
                /*reuse_a=*/false, /*reuse_b=*/false);
    }

    // output: one base pointer, offsets r*GROUPS*4 + nt*64 bytes (all < 2^23)
    float* oBase = out + (size_t)(mBase + half * 8) * GROUPS + nBase + l16;
    #pragma unroll
    for (int nt = 0; nt < 5; ++nt) {
        #pragma unroll
        for (int r = 0; r < 8; ++r) {
            __builtin_nontemporal_store(acc[nt][r] * INV_T,
                oBase + (size_t)r * GROUPS + nt * 16);
        }
    }
}

// ---------------------------------------------------------------------------
// Launch
// ---------------------------------------------------------------------------
extern "C" void kernel_launch(void* const* d_in, const int* in_sizes, int n_in,
                              void* d_out, int out_size, void* d_ws, size_t ws_size,
                              hipStream_t stream) {
    const float* x      = (const float*)d_in[0];  // [4096,128]
    // d_in[1] = x_index, d_in[2] = y : unused by the reference
    const float* memory = (const float*)d_in[3];  // [500000,128]
    const int*   gids   = (const int*)d_in[4];    // [500000]
    float* out = (float*)d_out;                   // [4096,10000]

    // workspace layout (8.77 MB total)
    char* ws = (char*)d_ws;
    float*          sums   = (float*)ws;                                  // G*D f32
    float*          counts = (float*)(ws + (size_t)GROUPS * DIM * 4);     // G   f32
    unsigned short* xb     = (unsigned short*)(ws + (size_t)GROUPS * DIM * 4
                                                  + (size_t)GROUPS * 4);  // B*D bf16
    unsigned short* vib    = (unsigned short*)((char*)xb
                                                  + (size_t)BATCH * DIM * 2); // G*D bf16

    // 1. zero accumulators
    {
        int n = GROUPS * DIM + GROUPS;
        zero_accum_kernel<<<(n + 255) / 256, 256, 0, stream>>>(sums, n);
    }
    // 2. segmented sum: one wave per row
    {
        segsum_kernel<<<NROWS / 8, 256, 0, stream>>>(memory, gids, sums, counts);
    }
    // 3. mean + normalize + bf16 pack vi: one wave per group
    {
        finalize_vi_kernel<<<GROUPS / 8, 256, 0, stream>>>(sums, counts, vib);
    }
    // 4. pack x to bf16
    {
        int n4 = BATCH * DIM / 4;
        cvt_x_kernel<<<(n4 + 255) / 256, 256, 0, stream>>>(x, xb, n4);
    }
    // 5. WMMA GEMM + scale
    {
        dim3 grid(625 / 5, BATCH / 16 / 8);    // (125, 32)
        gemm_wmma_kernel<<<grid, 256, 0, stream>>>(
            (const __bf16*)xb, (const __bf16*)vib, out);
    }
}
